// GNN_Model_19112604467556
// MI455X (gfx1250) — matrix operand order repaired and checked by reading, not compiled, not run on hardware
//
#include <hip/hip_runtime.h>
#include <hip/hip_bf16.h>

typedef _Float16 h16;
typedef __attribute__((ext_vector_type(16))) _Float16 v16h;
typedef __attribute__((ext_vector_type(8)))  _Float16 v8h;
typedef __attribute__((ext_vector_type(8)))  float    v8f;

#define NNODES 50000
#define NEDGES 800000
#define EPRIME (NEDGES + NNODES)   // edges + self loops
#define NQ     100000

// ---------------------------------------------------------------------------
// f32 -> f16 conversion
__global__ void cvt_h_kernel(const float* __restrict__ in, h16* __restrict__ out, long n) {
    long i = (long)blockIdx.x * blockDim.x + threadIdx.x;
    if (i < n) out[i] = (h16)in[i];
}

// W [K x N] f32 -> Wt [N x K] f16 (transposed so WMMA B-frag K-values are contiguous)
__global__ void transpose_cvt_kernel(const float* __restrict__ w, h16* __restrict__ wt,
                                     int K, int Nn) {
    int i = blockIdx.x * blockDim.x + threadIdx.x;
    if (i >= K * Nn) return;
    int k = i / Nn, n = i % Nn;
    wt[(size_t)n * K + k] = (h16)w[i];
}

__global__ void fill_f32_kernel(float* __restrict__ p, float v, long n) {
    long i = (long)blockIdx.x * blockDim.x + threadIdx.x;
    if (i < n) p[i] = v;
}

// ---------------------------------------------------------------------------
// WMMA GEMM: C[M x N] f32 = A[M x K] f16 (row major) * Bt[N x K] f16 (B transposed)
// One wave computes a 16 x (16*NT) strip: A fragment loaded once per K-step and
// reused across NT WMMAs (2.5 b128 loads per WMMA instead of 4).
// Guard is wave-uniform, so EXEC is all-ones for every wave reaching the WMMAs.
template <int NT>
__global__ __launch_bounds__(256) void gemm_wmma_kernel(const h16* __restrict__ A,
                                                        const h16* __restrict__ Bt,
                                                        float* __restrict__ C,
                                                        int M, int Nn, int K, int strips) {
    const int wave = (blockIdx.x * blockDim.x + threadIdx.x) >> 5;
    if (wave >= strips) return;                 // uniform per wave
    const int lane    = threadIdx.x & 31;
    const int stripsN = Nn / (16 * NT);
    const int tm      = wave / stripsN;
    const int ts      = wave % stripsN;
    const int hi      = lane >> 4;              // 0: lanes 0-15, 1: lanes 16-31
    const int l15     = lane & 15;

    // A frag: lane<16 -> row M=l15, K = kk..kk+15
    //         lane>=16 -> row M=l15, K = kk+16..kk+31
    const h16* aRow  = A  + (size_t)(tm * 16 + l15) * K + (hi ? 16 : 0);
    // B frag t: col N = ts*16*NT + t*16 + l15; lane<16 -> K=kk..kk+15, lane>=16 -> +16
    const h16* bBase = Bt + (size_t)(ts * 16 * NT + l15) * K + (hi ? 16 : 0);

    v8f acc[NT] = {};
    for (int kk = 0; kk < K; kk += 32) {
        union { v16h v; v8h h[2]; } a;
        a.h[0] = *(const v8h*)(aRow + kk);
        a.h[1] = *(const v8h*)(aRow + kk + 8);
#pragma unroll
        for (int t = 0; t < NT; ++t) {
            v16h b = *(const v16h*)(bBase + (size_t)t * 16 * K + kk);
            acc[t] = __builtin_amdgcn_wmma_f32_16x16x32_f16(
                /*neg_a=*/false, a.v, /*neg_b=*/false, b,
                /*c_mod=*/(short)0, acc[t], /*reuse_a=*/false, /*reuse_b=*/false);
        }
    }

    // C/D layout: lanes 0-15 VGPR r -> M=r; lanes 16-31 VGPR r -> M=8+r; N=l15
    float* cRow = C + (size_t)(tm * 16 + (hi ? 8 : 0)) * Nn + ts * 16 * NT + l15;
#pragma unroll
    for (int t = 0; t < NT; ++t)
#pragma unroll
        for (int r = 0; r < 8; ++r)
            cRow[(size_t)r * Nn + t * 16] = acc[t][r];
}

// ---------------------------------------------------------------------------
// Attention coefficients: out[n,h] = sum_c h[n,h,c] * a[h,c]
__global__ void attn_dot_kernel(const float* __restrict__ hmat, const float* __restrict__ avec,
                                float* __restrict__ out, int n, int H, int C) {
    int i = blockIdx.x * blockDim.x + threadIdx.x;
    if (i >= n * H) return;
    int node = i / H, hd = i % H;
    const float* hp = hmat + (size_t)node * H * C + (size_t)hd * C;
    const float* ap = avec + (size_t)hd * C;
    float s = 0.f;
    for (int c = 0; c < C; ++c) s += hp[c] * ap[c];
    out[i] = s;
}

__device__ __forceinline__ void edge_rc(const int* __restrict__ ei, int e, int& r, int& c) {
    if (e < NEDGES) { r = ei[e]; c = ei[NEDGES + e]; }
    else            { r = c = e - NEDGES; }          // self loop
}

__device__ __forceinline__ float leaky02(float x) { return x > 0.f ? x : 0.2f * x; }

// Pass 1: segment max of leaky-relu logits over destination node
__global__ void edge_max_kernel(const int* __restrict__ ei, const float* __restrict__ als,
                                const float* __restrict__ ald, float* __restrict__ m, int H) {
    int i = blockIdx.x * blockDim.x + threadIdx.x;
    if (i >= EPRIME * H) return;
    int e = i / H, hd = i % H, r, c;
    edge_rc(ei, e, r, c);
    float l = leaky02(als[r * H + hd] + ald[c * H + hd]);
    atomicMax(&m[c * H + hd], l);
}

// Pass 2: e = exp(logit - m[col]); segment sum
__global__ void edge_exp_kernel(const int* __restrict__ ei, const float* __restrict__ als,
                                const float* __restrict__ ald, const float* __restrict__ m,
                                float* __restrict__ s, float* __restrict__ expb, int H) {
    int i = blockIdx.x * blockDim.x + threadIdx.x;
    if (i >= EPRIME * H) return;
    int e = i / H, hd = i % H, r, c;
    edge_rc(ei, e, r, c);
    float l  = leaky02(als[r * H + hd] + ald[c * H + hd]);
    float ev = __expf(l - m[c * H + hd]);
    expb[i] = ev;
    atomicAdd(&s[c * H + hd], ev);
}

// Pass 3: out[col, h, c] += h[row, h, c] * (e / s[col, h])  — one block per edge,
// one thread per (head, channel); row gather is fully coalesced.
__global__ void edge_agg_kernel(const int* __restrict__ ei, const float* __restrict__ hmat,
                                const float* __restrict__ expb, const float* __restrict__ s,
                                float* __restrict__ out, int H, int C) {
    int e = blockIdx.x;
    int t = threadIdx.x;                 // t < H*C
    int hd = t / C;
    int r, c;
    edge_rc(ei, e, r, c);
    float alpha = expb[(size_t)e * H + hd] / s[c * H + hd];
    atomicAdd(&out[(size_t)c * H * C + t], hmat[(size_t)r * H * C + t] * alpha);
}

__global__ void bias_relu_h_kernel(const float* __restrict__ in, const float* __restrict__ b,
                                   h16* __restrict__ out, long n, int F) {
    long i = (long)blockIdx.x * blockDim.x + threadIdx.x;
    if (i >= n) return;
    float v = in[i] + b[i % F];
    out[i] = (h16)(v > 0.f ? v : 0.f);
}

__global__ void bias_add_kernel(const float* in, const float* __restrict__ b,
                                float* out, long n, int F) {   // supports in==out
    long i = (long)blockIdx.x * blockDim.x + threadIdx.x;
    if (i >= n) return;
    out[i] = in[i] + b[i % F];
}

// Link scores: one wave per query edge, 4 floats/lane, wave32 xor-reduction.
__global__ __launch_bounds__(256) void score_kernel(const float* __restrict__ z,
                                                    const int* __restrict__ eli,
                                                    float* __restrict__ out) {
    int wave = (blockIdx.x * blockDim.x + threadIdx.x) >> 5;
    int lane = threadIdx.x & 31;
    if (wave >= NQ) return;
    int a = eli[wave], b = eli[NQ + wave];
    float4 x = ((const float4*)(z + (size_t)a * 128))[lane];
    float4 y = ((const float4*)(z + (size_t)b * 128))[lane];
    float v = x.x * y.x + x.y * y.y + x.z * y.z + x.w * y.w;
#pragma unroll
    for (int off = 16; off > 0; off >>= 1) v += __shfl_xor(v, off, 32);
    if (lane == 0) out[wave] = v;
}

// ---------------------------------------------------------------------------
extern "C" void kernel_launch(void* const* d_in, const int* in_sizes, int n_in,
                              void* d_out, int out_size, void* d_ws, size_t ws_size,
                              hipStream_t stream) {
    (void)in_sizes; (void)n_in; (void)out_size; (void)ws_size;
    const float* x   = (const float*)d_in[0];
    const int*   ei  = (const int*)d_in[1];
    const int*   eli = (const int*)d_in[2];
    const float* W1  = (const float*)d_in[3];
    const float* as1 = (const float*)d_in[4];
    const float* ad1 = (const float*)d_in[5];
    const float* b1  = (const float*)d_in[6];
    const float* W2  = (const float*)d_in[7];
    const float* as2 = (const float*)d_in[8];
    const float* ad2 = (const float*)d_in[9];
    const float* b2  = (const float*)d_in[10];
    float* scores = (float*)d_out;

    char* ws = (char*)d_ws;
    size_t off = 0;
    auto alloc = [&](size_t bytes) -> void* {
        off = (off + 255) & ~(size_t)255;
        void* p = ws + off;
        off += bytes;
        return p;
    };

    h16*   xh   = (h16*)  alloc((size_t)NNODES * 128 * 2);
    h16*   w1t  = (h16*)  alloc((size_t)256 * 128 * 2);
    h16*   w2t  = (h16*)  alloc((size_t)128 * 256 * 2);
    float* h1   = (float*)alloc((size_t)NNODES * 256 * 4);   // reused as h2 (layer 2)
    float* als1 = (float*)alloc((size_t)NNODES * 4 * 4);     // reused layer 2
    float* ald1 = (float*)alloc((size_t)NNODES * 4 * 4);
    float* m1   = (float*)alloc((size_t)NNODES * 4 * 4);
    float* s1   = (float*)alloc((size_t)NNODES * 4 * 4);
    float* expb = (float*)alloc((size_t)EPRIME * 4 * 4);     // reused layer 2
    float* out1 = (float*)alloc((size_t)NNODES * 256 * 4);   // reused as out2/z
    h16*   h1h  = (h16*)  alloc((size_t)NNODES * 256 * 2);

    float* h2 = h1;          // layer-2 reuse
    float* als2 = als1, *ald2 = ald1, *m2 = m1, *s2 = s1;
    float* out2 = out1;

    const int T = 256;
    auto blk = [](long n, int t) { return (int)((n + t - 1) / t); };

    // ---- input conversion
    cvt_h_kernel<<<blk(6400000, T), T, 0, stream>>>(x, xh, 6400000L);
    transpose_cvt_kernel<<<blk(128 * 256, T), T, 0, stream>>>(W1, w1t, 128, 256);
    transpose_cvt_kernel<<<blk(256 * 128, T), T, 0, stream>>>(W2, w2t, 256, 128);

    // ---- layer 1: h1 = x @ W1   (3125 M-tiles x 4 N-strips of 64 = 12500 waves)
    {
        int strips = 3125 * (256 / 64);
        gemm_wmma_kernel<4><<<blk((long)strips * 32, T), T, 0, stream>>>(
            xh, w1t, h1, NNODES, 256, 128, strips);
    }
    attn_dot_kernel<<<blk(NNODES * 4, T), T, 0, stream>>>(h1, as1, als1, NNODES, 4, 64);
    attn_dot_kernel<<<blk(NNODES * 4, T), T, 0, stream>>>(h1, ad1, ald1, NNODES, 4, 64);
    fill_f32_kernel<<<blk(NNODES * 4, T), T, 0, stream>>>(m1, -3.0e38f, NNODES * 4L);
    fill_f32_kernel<<<blk(NNODES * 4, T), T, 0, stream>>>(s1, 0.f, NNODES * 4L);
    fill_f32_kernel<<<blk((long)NNODES * 256, T), T, 0, stream>>>(out1, 0.f, (long)NNODES * 256);
    edge_max_kernel<<<blk((long)EPRIME * 4, T), T, 0, stream>>>(ei, als1, ald1, m1, 4);
    edge_exp_kernel<<<blk((long)EPRIME * 4, T), T, 0, stream>>>(ei, als1, ald1, m1, s1, expb, 4);
    edge_agg_kernel<<<EPRIME, 256, 0, stream>>>(ei, h1, expb, s1, out1, 4, 64);
    bias_relu_h_kernel<<<blk((long)NNODES * 256, T), T, 0, stream>>>(out1, b1, h1h,
                                                                     (long)NNODES * 256, 256);

    // ---- layer 2: h2 = relu(h1) @ W2  (3125 M-tiles x 2 N-strips of 64 = 6250 waves)
    {
        int strips = 3125 * (128 / 64);
        gemm_wmma_kernel<4><<<blk((long)strips * 32, T), T, 0, stream>>>(
            h1h, w2t, h2, NNODES, 128, 256, strips);
    }
    attn_dot_kernel<<<blk(NNODES, T), T, 0, stream>>>(h2, as2, als2, NNODES, 1, 128);
    attn_dot_kernel<<<blk(NNODES, T), T, 0, stream>>>(h2, ad2, ald2, NNODES, 1, 128);
    fill_f32_kernel<<<blk(NNODES, T), T, 0, stream>>>(m2, -3.0e38f, (long)NNODES);
    fill_f32_kernel<<<blk(NNODES, T), T, 0, stream>>>(s2, 0.f, (long)NNODES);
    fill_f32_kernel<<<blk((long)NNODES * 128, T), T, 0, stream>>>(out2, 0.f, (long)NNODES * 128);
    edge_max_kernel<<<blk(EPRIME, T), T, 0, stream>>>(ei, als2, ald2, m2, 1);
    edge_exp_kernel<<<blk(EPRIME, T), T, 0, stream>>>(ei, als2, ald2, m2, s2, expb, 1);
    edge_agg_kernel<<<EPRIME, 128, 0, stream>>>(ei, h2, expb, s2, out2, 1, 128);
    bias_add_kernel<<<blk((long)NNODES * 128, T), T, 0, stream>>>(out2, b2, out2,
                                                                  (long)NNODES * 128, 128);

    // ---- decode: dot-product link scores (one wave per query)
    score_kernel<<<NQ / 8, 256, 0, stream>>>(out2, eli, scores);
}
